// Head_64141041598682
// MI455X (gfx1250) — matrix operand order repaired and checked by reading, not compiled, not run on hardware
//
#include <hip/hip_runtime.h>
#include <hip/hip_bf16.h>

typedef __attribute__((ext_vector_type(16))) _Float16 v16h;
typedef __attribute__((ext_vector_type(8)))  _Float16 v8h;
typedef __attribute__((ext_vector_type(8)))  float    v8f;

#define N_EMBD   1024
#define HEAD     128
#define BATCH    8
#define SEQ      2048
#define MTOT     (BATCH * SEQ)      // 16384 tokens

static __device__ __forceinline__ v16h pack16(v8h lo, v8h hi)
{
    v16h r;
    #pragma unroll
    for (int e = 0; e < 8; ++e) { r[e] = lo[e]; r[e + 8] = hi[e]; }
    return r;
}

// ---------------------------------------------------------------------------
// Kernel 0: one-shot transpose of W[1024,128] f32 -> Wt[128,1024] f16
// ---------------------------------------------------------------------------
__global__ __launch_bounds__(256) void transpose_w(const float* __restrict__ W,
                                                   _Float16* __restrict__ Wt)
{
    const int col   = threadIdx.x & 127;
    const int part  = threadIdx.x >> 7;              // 0/1
    const int kbase = blockIdx.x * 8 + part * 4;
    #pragma unroll
    for (int i = 0; i < 4; ++i) {
        int k = kbase + i;
        Wt[(size_t)col * N_EMBD + k] = (_Float16)W[(size_t)k * HEAD + col];
    }
}

// ---------------------------------------------------------------------------
// Kernel 1: projection  X[16384,1024] @ W[1024,128] -> out f16
// TR==false: out[token*128 + col]
// TR==true : out[b*128*2048 + col*2048 + s]   (V stored head-major)
// ---------------------------------------------------------------------------
#define STRX 40     // Xs row stride (halves), rows = tokens
#define STRWT 40    // Ws_t row stride (halves), rows = head cols

template <bool TR>
__global__ __launch_bounds__(256) void proj_kernel(const float* __restrict__ X,
                                                   const _Float16* __restrict__ Wt,
                                                   _Float16* __restrict__ out)
{
    __shared__ alignas(16) _Float16 Xs [128 * STRX];
    __shared__ alignas(16) _Float16 Wst[128 * STRWT];

    const int t    = threadIdx.x;
    const int lane = t & 31;
    const int wave = t >> 5;
    const int row0 = blockIdx.x * 128;

    const int ml = lane & 15;        // M (or N) index within 16
    const int hb = lane >> 4;        // half-wave select

    v8f acc[8];
    #pragma unroll
    for (int i = 0; i < 8; ++i)
        #pragma unroll
        for (int j = 0; j < 8; ++j) acc[i][j] = 0.0f;

    for (int k0 = 0; k0 < N_EMBD; k0 += 32) {
        // stage X tile: 128 rows x 32 cols, f32 -> f16, packed b64 stores
        #pragma unroll
        for (int i = 0; i < 4; ++i) {
            int lin = t + 256 * i;            // 0..1023
            int r   = lin >> 3;
            int c4  = lin & 7;
            float4 xv = *(const float4*)(X + (size_t)(row0 + r) * N_EMBD + k0 + c4 * 4);
            union { _Float16 h[4]; uint2 u; } pk;
            pk.h[0] = (_Float16)xv.x; pk.h[1] = (_Float16)xv.y;
            pk.h[2] = (_Float16)xv.z; pk.h[3] = (_Float16)xv.w;
            *(uint2*)&Xs[r * STRX + c4 * 4] = pk.u;
        }
        // stage W^T tile: 128 cols x 32 k, straight b128 copies from Wt
        #pragma unroll
        for (int i = 0; i < 2; ++i) {
            int lin = t + 256 * i;            // 0..511
            int r   = lin >> 2;               // head col 0..127
            int c8  = lin & 3;                // k group
            const v8h* src = (const v8h*)(Wt + (size_t)r * N_EMBD + k0 + c8 * 8);
            *(v8h*)&Wst[r * STRWT + c8 * 8] = *src;
        }
        __syncthreads();

        // A fragment (16x32 f16): lane ml = row, hb selects K half
        const int m  = (wave << 4) + ml;
        const int kb = hb << 3;                        // 0 or 8
        v16h a = pack16(*(const v8h*)&Xs[m * STRX + kb],
                        *(const v8h*)&Xs[m * STRX + kb + 16]);

        const int kr = hb << 3;                        // 0 or 8
        // grouped B-fragment loads: 4 frags in flight per wait, then 4 WMMAs
        #pragma unroll
        for (int g = 0; g < 2; ++g) {
            v16h bq[4];
            #pragma unroll
            for (int u = 0; u < 4; ++u) {
                const _Float16* p = &Wst[((g * 4 + u) * 16 + ml) * STRWT + kr];
                bq[u] = pack16(*(const v8h*)p, *(const v8h*)(p + 16));
            }
            #pragma unroll
            for (int u = 0; u < 4; ++u)
                acc[g * 4 + u] = __builtin_amdgcn_wmma_f32_16x16x32_f16(
                    false, a, false, bq[u], (short)0, acc[g * 4 + u], false, false);
        }
        __syncthreads();
    }

    // write out (f16)
    #pragma unroll
    for (int tt = 0; tt < 8; ++tt)
        #pragma unroll
        for (int r = 0; r < 8; ++r) {
            int gm = row0 + (wave << 4) + hb * 8 + r;  // token
            int gc = tt * 16 + ml;                     // head col
            _Float16 val = (_Float16)acc[tt][r];
            if (!TR) {
                out[(size_t)gm * HEAD + gc] = val;
            } else {
                int bb = gm >> 11;
                int s  = gm & 2047;
                out[((size_t)bb * HEAD + gc) * SEQ + s] = val;
            }
        }
}

// ---------------------------------------------------------------------------
// Kernel 2: attention with exact double softmax, 2-pass flash style.
//   pass 1: online (rowmax m, rowsum l) of exp(scores)
//   pass 2: recompute scores, p = exp(exp(s-m)/l); l2 += p; acc += p*V; out = acc/l2
// ---------------------------------------------------------------------------
#define SKS 136   // Ks row stride (halves), rows = keys
#define SVT 72    // Vt row stride (halves), rows = head dims
#define SPS 40    // prob staging row stride (halves), rows = 16 query rows
#define CHUNK 64

__global__ __launch_bounds__(256) void attn_kernel(const _Float16* __restrict__ qh,
                                                   const _Float16* __restrict__ kh,
                                                   const _Float16* __restrict__ vhT,
                                                   float* __restrict__ out)
{
    __shared__ alignas(16) _Float16 Ks[CHUNK * SKS];
    __shared__ alignas(16) _Float16 Vt[HEAD * SVT];
    __shared__ alignas(16) _Float16 Ps[8][16 * SPS];   // per-wave, A-layout staging

    const int t    = threadIdx.x;
    const int lane = t & 31;
    const int wave = t >> 5;
    const int b    = blockIdx.y;
    const int q0   = blockIdx.x * 128 + wave * 16;   // query row within batch

    const int ml = lane & 15;
    const int hb = lane >> 4;

    const float scale = 0.08838834764831845f;        // 1/sqrt(128)

    // --- load Q fragments (kept in registers for both passes) ---
    v16h qf[4];
    {
        const size_t qtok = (size_t)b * SEQ + q0 + ml;
        #pragma unroll
        for (int f = 0; f < 4; ++f) {
            const _Float16* p = qh + qtok * HEAD + f * 32 + hb * 8;
            qf[f] = pack16(*(const v8h*)p, *(const v8h*)(p + 16));
        }
    }

    float rowM[8], rowL[8];
    #pragma unroll
    for (int r = 0; r < 8; ++r) { rowM[r] = -3.0e38f; rowL[r] = 0.0f; }

    // =============================== PASS 1 ===============================
    for (int c0 = 0; c0 < SEQ; c0 += CHUNK) {
        // stage K chunk: 64 keys x 128 head, b128 loads
        #pragma unroll
        for (int i = 0; i < 4; ++i) {
            int lin = t + 256 * i;
            int r   = lin >> 4;
            int c8  = lin & 15;
            const v8h* src = (const v8h*)(kh + ((size_t)b * SEQ + c0 + r) * HEAD + c8 * 8);
            *(v8h*)&Ks[r * SKS + c8 * 8] = *src;
        }
        if (c0 + CHUNK < SEQ)   // prefetch next K chunk (global_prefetch_b8)
            __builtin_prefetch(kh + ((size_t)b * SEQ + c0 + CHUNK + (t >> 2)) * HEAD, 0, 1);
        __syncthreads();

        #pragma unroll
        for (int sub = 0; sub < 4; ++sub) {
            // batch all 4 B fragments, then 4 chained WMMAs
            v16h bq[4];
            #pragma unroll
            for (int f = 0; f < 4; ++f) {
                const _Float16* p = &Ks[(sub * 16 + ml) * SKS + f * 32 + hb * 8];
                bq[f] = pack16(*(const v8h*)p, *(const v8h*)(p + 16));
            }
            v8f c;
            #pragma unroll
            for (int j = 0; j < 8; ++j) c[j] = 0.0f;
            #pragma unroll
            for (int f = 0; f < 4; ++f)
                c = __builtin_amdgcn_wmma_f32_16x16x32_f16(
                    false, qf[f], false, bq[f], (short)0, c, false, false);

            // online (max, sumexp) per query row; rows live in half-waves
            #pragma unroll
            for (int r = 0; r < 8; ++r) {
                float s = c[r] * scale;
                float v = s;
                #pragma unroll
                for (int off = 1; off < 16; off <<= 1)
                    v = fmaxf(v, __shfl_xor(v, off, 32));
                float mnew = fmaxf(rowM[r], v);
                float ev = __expf(s - mnew);
                #pragma unroll
                for (int off = 1; off < 16; off <<= 1)
                    ev += __shfl_xor(ev, off, 32);
                rowL[r] = rowL[r] * __expf(rowM[r] - mnew) + ev;
                rowM[r] = mnew;
            }
        }
        __syncthreads();
    }

    // =============================== PASS 2 ===============================
    float l2[8];
    v8f acc[8];
    #pragma unroll
    for (int r = 0; r < 8; ++r) l2[r] = 0.0f;
    #pragma unroll
    for (int i = 0; i < 8; ++i)
        #pragma unroll
        for (int j = 0; j < 8; ++j) acc[i][j] = 0.0f;

    float rinv[8];
    #pragma unroll
    for (int r = 0; r < 8; ++r) rinv[r] = 1.0f / rowL[r];

    for (int c0 = 0; c0 < SEQ; c0 += CHUNK) {
        #pragma unroll
        for (int i = 0; i < 4; ++i) {               // K chunk
            int lin = t + 256 * i;
            int r   = lin >> 4;
            int c8  = lin & 15;
            const v8h* src = (const v8h*)(kh + ((size_t)b * SEQ + c0 + r) * HEAD + c8 * 8);
            *(v8h*)&Ks[r * SKS + c8 * 8] = *src;
        }
        #pragma unroll
        for (int i = 0; i < 4; ++i) {               // V chunk (head-major)
            int lin = t + 256 * i;
            int r   = lin >> 3;                     // head dim 0..127
            int c8  = lin & 7;                      // key group
            const v8h* src = (const v8h*)(vhT + ((size_t)b * HEAD + r) * SEQ + c0 + c8 * 8);
            *(v8h*)&Vt[r * SVT + c8 * 8] = *src;
        }
        __syncthreads();

        #pragma unroll
        for (int pair = 0; pair < 2; ++pair) {
            // scores + double-softmax probabilities for 32 keys -> Ps (A-layout)
            #pragma unroll
            for (int sub = 0; sub < 2; ++sub) {
                v16h bq[4];
                #pragma unroll
                for (int f = 0; f < 4; ++f) {
                    const _Float16* p = &Ks[(pair * 32 + sub * 16 + ml) * SKS + f * 32 + hb * 8];
                    bq[f] = pack16(*(const v8h*)p, *(const v8h*)(p + 16));
                }
                v8f c;
                #pragma unroll
                for (int j = 0; j < 8; ++j) c[j] = 0.0f;
                #pragma unroll
                for (int f = 0; f < 4; ++f)
                    c = __builtin_amdgcn_wmma_f32_16x16x32_f16(
                        false, qf[f], false, bq[f], (short)0, c, false, false);

                // Ps[m][key]: strided independent b16 stores (no wait chains)
                #pragma unroll
                for (int r = 0; r < 8; ++r) {
                    float w1 = __expf(c[r] * scale - rowM[r]) * rinv[r]; // softmax #1
                    float p2 = __expf(w1);                               // softmax #2 (unnorm)
                    l2[r] += p2;                                         // lane-partial sum
                    Ps[wave][(hb * 8 + r) * SPS + sub * 16 + ml] = (_Float16)p2;
                }
            }
            // A fragment: contiguous b128 reads from Ps (A-layout)
            v16h a;
            {
                const _Float16* p = &Ps[wave][ml * SPS + hb * 8];
                a = pack16(*(const v8h*)p, *(const v8h*)(p + 16));
            }
            // acc += P(16x32) x V(32x128); groups of 2 to bound VGPR pressure
            #pragma unroll
            for (int g = 0; g < 4; ++g) {
                v16h bq[2];
                #pragma unroll
                for (int u = 0; u < 2; ++u) {
                    const _Float16* p = &Vt[((g * 2 + u) * 16 + ml) * SVT + pair * 32 + hb * 8];
                    bq[u] = pack16(*(const v8h*)p, *(const v8h*)(p + 16));
                }
                #pragma unroll
                for (int u = 0; u < 2; ++u)
                    acc[g * 2 + u] = __builtin_amdgcn_wmma_f32_16x16x32_f16(
                        false, a, false, bq[u], (short)0, acc[g * 2 + u], false, false);
            }
        }
        __syncthreads();
    }

    // finish l2 row sums across the half-wave, then normalize + store f32
    #pragma unroll
    for (int r = 0; r < 8; ++r) {
        float v = l2[r];
        #pragma unroll
        for (int off = 1; off < 16; off <<= 1)
            v += __shfl_xor(v, off, 32);
        l2[r] = 1.0f / v;
    }
    #pragma unroll
    for (int tt = 0; tt < 8; ++tt)
        #pragma unroll
        for (int r = 0; r < 8; ++r) {
            size_t row = (size_t)b * SEQ + q0 + hb * 8 + r;
            out[row * HEAD + tt * 16 + ml] = acc[tt][r] * l2[r];
        }
}

// ---------------------------------------------------------------------------
extern "C" void kernel_launch(void* const* d_in, const int* in_sizes, int n_in,
                              void* d_out, int out_size, void* d_ws, size_t ws_size,
                              hipStream_t stream)
{
    const float* q  = (const float*)d_in[0];
    const float* k  = (const float*)d_in[1];
    const float* v  = (const float*)d_in[2];
    const float* Wq = (const float*)d_in[3];
    const float* Wk = (const float*)d_in[4];
    const float* Wv = (const float*)d_in[5];

    _Float16* qh  = (_Float16*)d_ws;
    _Float16* kh  = qh  + (size_t)MTOT * HEAD;
    _Float16* vhT = kh  + (size_t)MTOT * HEAD;
    _Float16* WqT = vhT + (size_t)MTOT * HEAD;
    _Float16* WkT = WqT + (size_t)HEAD * N_EMBD;
    _Float16* WvT = WkT + (size_t)HEAD * N_EMBD;

    transpose_w<<<N_EMBD / 8, 256, 0, stream>>>(Wq, WqT);
    transpose_w<<<N_EMBD / 8, 256, 0, stream>>>(Wk, WkT);
    transpose_w<<<N_EMBD / 8, 256, 0, stream>>>(Wv, WvT);

    proj_kernel<false><<<MTOT / 128, 256, 0, stream>>>(q, WqT, qh);
    proj_kernel<false><<<MTOT / 128, 256, 0, stream>>>(k, WkT, kh);
    proj_kernel<true ><<<MTOT / 128, 256, 0, stream>>>(v, WvT, vhT);

    attn_kernel<<<dim3(SEQ / 128, BATCH), 256, 0, stream>>>(qh, kh, vhT, (float*)d_out);
}
